// SpectralConv2d_DDP_38671885533645
// MI455X (gfx1250) — matrix operand order repaired and checked by reading, not compiled, not run on hardware
//
#include <hip/hip_runtime.h>
#include <math.h>

typedef __attribute__((ext_vector_type(16))) _Float16 v16h;
typedef __attribute__((ext_vector_type(8)))  float    v8f;

#define B_    16
#define CI    64
#define CO    64
#define H_    256
#define W_    256
#define M1_   16
#define M2_   16
#define KX_   32
#define NMODE 512
#define PI2   6.283185307179586f

// ---- WMMA f16 operand index maps (per CDNA5 ISA 7.12.2, wave32) ----
__device__ __forceinline__ int a_m(int lane) { return lane & 15; }
__device__ __forceinline__ int a_k(int lane, int e) {
  int k = (e < 8) ? e : (e + 8);          // lane<16: K = 0..7, 16..23
  return (lane < 16) ? k : (k + 8);       // lane>=16: K = 8..15, 24..31
}
__device__ __forceinline__ int b_n(int lane) { return lane & 15; }
__device__ __forceinline__ int b_k(int lane, int e) { return a_k(lane, e); }

__device__ __forceinline__ v8f wmma16(v16h a, v16h b, v8f c) {
  return __builtin_amdgcn_wmma_f32_16x16x32_f16(false, a, false, b, (short)0, c, false, false);
}

// =====================================================================
// Kernel 1: per (b,ci) slab — forward DFT over w (16 modes) then over h
// (32 modes), fully fused through LDS.  Writes X[mode][re/im][b][ci] f16.
// =====================================================================
__global__ void __launch_bounds__(256) fno_fwd(const float* __restrict__ x,
                                               _Float16* __restrict__ Xws)
{
  extern __shared__ char smem[];
  _Float16* xh  = (_Float16*)smem;          // 128 x 260 (padded)  f16 slab half
  _Float16* tBc = xh  + 128 * 260;          // 8*32*16  stage-1 B tiles: cos
  _Float16* tBs = tBc + 8 * 32 * 16;        // 8*32*16  stage-1 B tiles: -sin
  _Float16* tAc = tBs + 8 * 32 * 16;        // 2*8*32*16 stage-2 A tiles: cos
  _Float16* tAs = tAc + 2 * 8 * 32 * 16;    // 2*8*32*16 stage-2 A tiles: sin
  _Float16* Gre = tAs + 2 * 8 * 32 * 16;    // 256*16
  _Float16* Gim = Gre + 256 * 16;           // 256*16

  const int tid  = threadIdx.x;
  const int lane = tid & 31;
  const int wave = tid >> 5;
  const int bi   = blockIdx.x;              // b*64 + ci
  const int b    = bi >> 6;
  const int ci   = bi & 63;

  // ---- twiddle tables, pre-laid-out in WMMA register order ----
  for (int idx = tid; idx < 8 * 32 * 16; idx += 256) {
    int kb = idx >> 9, l = (idx >> 4) & 31, e = idx & 15;
    int w  = kb * 32 + b_k(l, e);
    int ky = l & 15;
    float th = PI2 * (float)((ky * w) & 255) / 256.0f;
    float s, c; __sincosf(th, &s, &c);
    tBc[idx] = (_Float16)c;                 // Gre += x*cos
    tBs[idx] = (_Float16)(-s);              // Gim += x*(-sin)
  }
  for (int idx = tid; idx < 2 * 8 * 32 * 16; idx += 256) {
    int t = idx >> 12, rem = idx & 4095;
    int kb = rem >> 9, l = (rem >> 4) & 31, e = rem & 15;
    int kx = (t == 0) ? (l & 15) : (240 + (l & 15));
    int h  = kb * 32 + a_k(l, e);
    float th = PI2 * (float)((kx * h) & 255) / 256.0f;
    float s, c; __sincosf(th, &s, &c);
    tAc[idx] = (_Float16)c;
    tAs[idx] = (_Float16)s;
  }

  const float* xs = x + (size_t)bi * (H_ * W_);

  for (int half = 0; half < 2; ++half) {
    __syncthreads();                        // protect xh before overwrite
    { // coalesced f32 load -> f16 LDS, 128 rows
      int r  = tid >> 1;
      int cb = (tid & 1) * 128;
      const float* src = xs + (size_t)(half * 128 + r) * W_ + cb;
      _Float16* dst = xh + r * 260 + cb;
#pragma unroll 8
      for (int j = 0; j < 128; j += 4) {
        float4 v = *(const float4*)(src + j);
        dst[j + 0] = (_Float16)v.x; dst[j + 1] = (_Float16)v.y;
        dst[j + 2] = (_Float16)v.z; dst[j + 3] = (_Float16)v.w;
      }
    }
    __syncthreads();
    { // GEMM1: wave -> one 16-row h tile of this half
      v8f are = {}; v8f aim = {};
#pragma unroll
      for (int kb = 0; kb < 8; ++kb) {
        v16h A, Bc, Bs;
#pragma unroll
        for (int e = 0; e < 16; ++e) {
          int hl = wave * 16 + a_m(lane);
          int w  = kb * 32 + a_k(lane, e);
          A[e]  = xh[hl * 260 + w];
          Bc[e] = tBc[(kb * 32 + lane) * 16 + e];
          Bs[e] = tBs[(kb * 32 + lane) * 16 + e];
        }
        are = wmma16(A, Bc, are);
        aim = wmma16(A, Bs, aim);
      }
      int ky = lane & 15;
      int mo = (lane < 16) ? 0 : 8;
#pragma unroll
      for (int r = 0; r < 8; ++r) {
        int h = half * 128 + wave * 16 + r + mo;
        Gre[h * 16 + ky] = (_Float16)are[r];
        Gim[h * 16 + ky] = (_Float16)aim[r];
      }
    }
  }
  __syncthreads();

  // GEMM2: X = e^{-i*2pi*kx*h/H} . G  ; waves 0..3 = (kxtile, re/im)
  if (wave < 4) {
    int kxt   = wave >> 1;
    int which = wave & 1;                   // 0: Xre = c*Gre + s*Gim
    v8f acc = {};                           // 1: Xim = c*Gim + s*(-Gre)
#pragma unroll
    for (int kb = 0; kb < 8; ++kb) {
      v16h Ac, As, B1, B2;
#pragma unroll
      for (int e = 0; e < 16; ++e) {
        int tix = ((kxt * 8 + kb) * 32 + lane) * 16 + e;
        Ac[e] = tAc[tix];
        As[e] = tAs[tix];
        int h  = kb * 32 + b_k(lane, e);
        int ky = lane & 15;
        _Float16 gre = Gre[h * 16 + ky];
        _Float16 gim = Gim[h * 16 + ky];
        if (which == 0) { B1[e] = gre; B2[e] = gim; }
        else            { B1[e] = gim; B2[e] = (_Float16)(-gre); }
      }
      acc = wmma16(As, B2, acc);
      acc = wmma16(Ac, B1, acc);
    }
    int ky = lane & 15;
    int mo = (lane < 16) ? 0 : 8;
#pragma unroll
    for (int r = 0; r < 8; ++r) {
      int kx   = kxt * 16 + r + mo;
      int mode = kx * 16 + ky;
      Xws[((size_t)(mode * 2 + which) * 16 + b) * 64 + ci] = (_Float16)acc[r];
    }
  }
}

// =====================================================================
// Kernel 2: per mode — complex channel mix  O[b,o] = sum_i X[b,i]*W[i,o]
// 4 real 16x64x64 GEMMs (signs folded into B since f16 WMMA lacks NEG).
// =====================================================================
__global__ void __launch_bounds__(256) fno_mix(const _Float16* __restrict__ Xws,
                                               const float* __restrict__ w1,
                                               const float* __restrict__ w2,
                                               _Float16* __restrict__ Ows)
{
  const int tid  = threadIdx.x;
  const int lane = tid & 31;
  const int wave = tid >> 5;
  const int mode = blockIdx.x;
  const int kxi  = mode >> 4;
  const int ky   = mode & 15;
  const float* wgt = (kxi < 16) ? w1 : w2;
  const int mx     = kxi & 15;

  const int which = wave & 1;               // 0: Ore, 1: Oim
  const int nt    = wave >> 1;              // Co tile (4 of 16)

  v8f acc = {};
#pragma unroll
  for (int kb = 0; kb < 2; ++kb) {
    v16h Are, Aim, B1, B2;
#pragma unroll
    for (int e = 0; e < 16; ++e) {
      int i  = kb * 32 + a_k(lane, e);
      int bb = a_m(lane);
      Are[e] = Xws[((size_t)(mode * 2 + 0) * 16 + bb) * 64 + i];
      Aim[e] = Xws[((size_t)(mode * 2 + 1) * 16 + bb) * 64 + i];
    }
#pragma unroll
    for (int e = 0; e < 16; ++e) {
      int i = kb * 32 + b_k(lane, e);
      int o = nt * 16 + (lane & 15);
      const float* p = wgt + (size_t)((((i * CO + o) * M1_ + mx) * M2_ + ky) << 1);
      float wre = p[0], wim = p[1];
      if (which == 0) { B1[e] = (_Float16)wre;  B2[e] = (_Float16)(-wim); }
      else            { B1[e] = (_Float16)wim;  B2[e] = (_Float16)wre;   }
    }
    acc = wmma16(Are, B1, acc);             // Xre*Wre   | Xre*Wim
    acc = wmma16(Aim, B2, acc);             // Xim*(-Wim)| Xim*Wre
  }
  int o  = nt * 16 + (lane & 15);
  int mo = (lane < 16) ? 0 : 8;
#pragma unroll
  for (int r = 0; r < 8; ++r) {
    int bb = r + mo;
    Ows[(((size_t)(bb * CO + o) * 2 + which) * KX_ + kxi) * M2_ + ky] = (_Float16)acc[r];
  }
}

// =====================================================================
// Kernel 3: per (b,o) — inverse DFT over kx (32 modes -> 256 h) into LDS,
// then real inverse over w with Hermitian 2x factors; writes [256,256] f32.
// =====================================================================
__global__ void __launch_bounds__(256) fno_inv(const _Float16* __restrict__ Ows,
                                               float* __restrict__ out)
{
  extern __shared__ char smem[];
  _Float16* tAc = (_Float16*)smem;          // 16*32*16
  _Float16* tAs = tAc + 16 * 32 * 16;       // 16*32*16
  _Float16* tB3 = tAs + 16 * 32 * 16;       // 16*32*16 (cos/-sin, *2, /256)
  _Float16* zl  = tB3 + 16 * 32 * 16;       // 256 x 34 (padded)

  const int tid  = threadIdx.x;
  const int lane = tid & 31;
  const int wave = tid >> 5;
  const int bo   = blockIdx.x;              // b*64 + o

  for (int idx = tid; idx < 16 * 32 * 16; idx += 256) {
    int ht = idx >> 9, l = (idx >> 4) & 31, e = idx & 15;
    int h  = ht * 16 + (l & 15);
    int k  = a_k(l, e);
    int kx = (k < 16) ? k : (224 + k);      // 16..31 -> 240..255
    float th = PI2 * (float)((kx * h) & 255) / 256.0f;
    float s, c; __sincosf(th, &s, &c);
    tAc[idx] = (_Float16)c;
    tAs[idx] = (_Float16)s;
  }
  for (int idx = tid; idx < 16 * 32 * 16; idx += 256) {
    int wt = idx >> 9, l = (idx >> 4) & 31, e = idx & 15;
    int w  = wt * 16 + (l & 15);
    int k  = b_k(l, e);
    float v;
    if (k < 16) { int ky = k;
      v = (ky == 0) ? 1.0f :  2.0f * cosf(PI2 * (float)((ky * w) & 255) / 256.0f);
    } else {      int ky = k - 16;
      v = (ky == 0) ? 0.0f : -2.0f * sinf(PI2 * (float)((ky * w) & 255) / 256.0f);
    }
    tB3[idx] = (_Float16)(v * (1.0f / 256.0f));
  }
  __syncthreads();

  // O mode tiles (each wave keeps its own copy in registers)
  v16h Ore, Oim, OimN;
#pragma unroll
  for (int e = 0; e < 16; ++e) {
    int kx = b_k(lane, e);
    int ky = lane & 15;
    _Float16 ore = Ows[(((size_t)bo * 2 + 0) * KX_ + kx) * M2_ + ky];
    _Float16 oim = Ows[(((size_t)bo * 2 + 1) * KX_ + kx) * M2_ + ky];
    Ore[e] = ore; Oim[e] = oim; OimN[e] = (_Float16)(-oim);
  }
  // inverse over kx: z = e^{+i*2pi*kx*h/H} . O   (scaled by 1/256)
#pragma unroll
  for (int hti = 0; hti < 2; ++hti) {
    int ht = wave + hti * 8;
    v16h Ac, As;
#pragma unroll
    for (int e = 0; e < 16; ++e) {
      int tix = (ht * 32 + lane) * 16 + e;
      Ac[e] = tAc[tix]; As[e] = tAs[tix];
    }
    v8f zre = {}; v8f zim = {};
    zre = wmma16(As, OimN, zre);            // -s*Oim
    zre = wmma16(Ac, Ore,  zre);            // +c*Ore
    zim = wmma16(As, Ore,  zim);            // +s*Ore
    zim = wmma16(Ac, Oim,  zim);            // +c*Oim
    int ky = lane & 15;
    int mo = (lane < 16) ? 0 : 8;
#pragma unroll
    for (int r = 0; r < 8; ++r) {
      int h = ht * 16 + r + mo;
      zl[h * 34 + ky]      = (_Float16)(zre[r] * (1.0f / 256.0f));
      zl[h * 34 + 16 + ky] = (_Float16)(zim[r] * (1.0f / 256.0f));
    }
  }
  __syncthreads();

  // inverse over w: out[h,w] = zre0 + 2*sum(zre*cos - zim*sin)   (K=32)
  float* obase = out + (size_t)bo * (H_ * W_);
  for (int t = wave; t < 256; t += 8) {
    int ht = t >> 4, wt = t & 15;
    v16h A, Bt;
#pragma unroll
    for (int e = 0; e < 16; ++e) {
      int h = ht * 16 + a_m(lane);
      A[e]  = zl[h * 34 + a_k(lane, e)];
      Bt[e] = tB3[(wt * 32 + lane) * 16 + e];
    }
    v8f d = {};
    d = wmma16(A, Bt, d);
    int wc = wt * 16 + (lane & 15);
    int mo = (lane < 16) ? 0 : 8;
#pragma unroll
    for (int r = 0; r < 8; ++r) {
      int h = ht * 16 + r + mo;
      obase[(size_t)h * W_ + wc] = d[r];
    }
  }
}

// =====================================================================
extern "C" void kernel_launch(void* const* d_in, const int* in_sizes, int n_in,
                              void* d_out, int out_size, void* d_ws, size_t ws_size,
                              hipStream_t stream)
{
  (void)in_sizes; (void)n_in; (void)out_size; (void)ws_size;
  const float* x  = (const float*)d_in[0];
  const float* w1 = (const float*)d_in[1];
  const float* w2 = (const float*)d_in[2];
  float* out = (float*)d_out;

  _Float16* Xws = (_Float16*)d_ws;                       // 512*2*16*64 f16 = 2 MB
  _Float16* Ows = Xws + (size_t)NMODE * 2 * 16 * 64;     // 16*64*2*32*16 f16 = 2 MB

  const size_t lds_fwd = (size_t)(128 * 260 + 2 * (8 * 32 * 16) +
                                  2 * (2 * 8 * 32 * 16) + 2 * (256 * 16)) * sizeof(_Float16);
  const size_t lds_inv = (size_t)(3 * (16 * 32 * 16) + 256 * 34) * sizeof(_Float16);

  (void)hipFuncSetAttribute(reinterpret_cast<const void*>(fno_fwd),
                            hipFuncAttributeMaxDynamicSharedMemorySize, (int)lds_fwd);
  (void)hipFuncSetAttribute(reinterpret_cast<const void*>(fno_inv),
                            hipFuncAttributeMaxDynamicSharedMemorySize, (int)lds_inv);

  fno_fwd<<<B_ * CI, 256, lds_fwd, stream>>>(x, Xws);
  fno_mix<<<NMODE, 256, 0, stream>>>(Xws, w1, w2, Ows);
  fno_inv<<<B_ * CO, 256, lds_inv, stream>>>(Ows, out);
}
